// ContinuousDiffusion_30872224924148
// MI455X (gfx1250) — matrix operand order repaired and checked by reading, not compiled, not run on hardware
//
#include <hip/hip_runtime.h>
#include <hip/hip_bf16.h>
#include <math.h>

// ---------------------------------------------------------------------------
// ContinuousDiffusion graph-ODE for MI455X (gfx1250, wave32, WMMA).
// h = tanh(gat @ W_in + b_in); 12x { agg = segmean(h); h += (tanh(gelu(agg@W1)@W2) - decay*h)*dt }
// All GEMMs via v_wmma_f32_16x16x32_f16 (f16 operands, f32 accumulate).
// ---------------------------------------------------------------------------

typedef __attribute__((ext_vector_type(16))) _Float16 v16h;
typedef __attribute__((ext_vector_type(8)))  _Float16 v8h;
typedef __attribute__((ext_vector_type(8)))  float    v8f;

#define HID   128
#define HID2  256
#define DT    0.5f   /* TIME/STEPS = 6.0/12 */
#define NSTEP 12

__device__ __forceinline__ v8f wmma_f16(v16h a, v16h b, v8f c) {
  return __builtin_amdgcn_wmma_f32_16x16x32_f16(false, a, false, b, (short)0, c, false, false);
}

__device__ __forceinline__ v16h combine16(v8h lo, v8h hi) {
  v16h r;
#pragma unroll
  for (int i = 0; i < 8; ++i) { r[i] = lo[i]; r[8 + i] = hi[i]; }
  return r;
}

// A-fragment (16x32, f16) of a row-major [*, ld] f16 matrix.
// lanes 0-15: M=lane, K = k0+{0..7, 16..23}; lanes 16-31: M=lane-16, K = k0+{8..15, 24..31}.
__device__ __forceinline__ v16h load_a_frag(const _Float16* base, int ld, int row_base,
                                            int k0, int lane, int max_row) {
  int m = row_base + (lane & 15);
  if (m > max_row) m = max_row;           // clamp tail rows (stores are masked later)
  int kb = k0 + ((lane >> 4) << 3);       // + (lane/16)*8
  const _Float16* p = base + (size_t)m * ld + kb;
  v8h lo = *(const v8h*)(p);
  v8h hi = *(const v8h*)(p + 16);
  return combine16(lo, hi);
}

// ---------------------------------------------------------------------------
// Prep kernels
// ---------------------------------------------------------------------------

// Repack a row-major f32 weight [K, Ncols] into f16 B-fragment order:
// Wf[((nt*(K/32)+kt)*32 + lane)*16 + e] = W[kt*32 + (lane/16)*8 + (e%8) + (e/8)*16][nt*16 + lane%16]
__global__ void prep_weights_k(const float* __restrict__ W, _Float16* __restrict__ Wf,
                               int K, int Ncols) {
  int idx = blockIdx.x * blockDim.x + threadIdx.x;
  int total = K * Ncols;
  if (idx >= total) return;
  int e    = idx & 15;
  int lane = (idx >> 4) & 31;
  int kt   = (idx >> 9) % (K >> 5);
  int nt   = (idx >> 9) / (K >> 5);
  int n = nt * 16 + (lane & 15);
  int k = kt * 32 + ((lane >> 4) << 3) + (e & 7) + ((e >> 3) << 4);
  Wf[idx] = (_Float16)W[(size_t)k * Ncols + n];
}

__global__ void cvt_f32_to_f16_k(const float* __restrict__ in, _Float16* __restrict__ out, int n) {
  int i = blockIdx.x * blockDim.x + threadIdx.x;
  if (i < n) out[i] = (_Float16)in[i];
}

__global__ void zero_f32_k(float* __restrict__ p, int n) {
  int i = blockIdx.x * blockDim.x + threadIdx.x;
  if (i < n) p[i] = 0.0f;
}

__global__ void deg_k(const int* __restrict__ dst, float* __restrict__ deg, int E) {
  int e = blockIdx.x * blockDim.x + threadIdx.x;
  if (e < E) atomicAdd(&deg[dst[e]], 1.0f);
}

__global__ void invdeg_k(float* __restrict__ deg, int n) {
  int i = blockIdx.x * blockDim.x + threadIdx.x;
  if (i < n) deg[i] = 1.0f / fmaxf(deg[i], 1.0f);
}

// ---------------------------------------------------------------------------
// Edge scatter: 32 threads per edge, float4 per thread (gather hits L2; f32
// atomic-add-without-return resolves at the L2 atomic units).
// ---------------------------------------------------------------------------
__global__ void scatter_k(const float* __restrict__ h, const int* __restrict__ src,
                          const int* __restrict__ dst, float* __restrict__ agg, int E) {
  int t = blockIdx.x * blockDim.x + threadIdx.x;
  int e = t >> 5;
  if (e >= E) return;
  int c = (t & 31) << 2;
  int s = src[e], d = dst[e];
  const float4 v = *(const float4*)(h + (size_t)s * HID + c);
  float* o = agg + (size_t)d * HID + c;
  atomicAdd(o + 0, v.x);
  atomicAdd(o + 1, v.y);
  atomicAdd(o + 2, v.z);
  atomicAdd(o + 3, v.w);
}

__global__ void scale_cvt_k(const float* __restrict__ agg, const float* __restrict__ invdeg,
                            _Float16* __restrict__ aggh, int total) {
  int i = blockIdx.x * blockDim.x + threadIdx.x;
  if (i < total) aggh[i] = (_Float16)(agg[i] * invdeg[i >> 7]);
}

// ---------------------------------------------------------------------------
// Init GEMM: h = tanh(gat_f16 @ W_in + b_in).  8 waves/block, 16 rows/wave.
// ---------------------------------------------------------------------------
__global__ void __launch_bounds__(256)
init_gemm_k(const _Float16* __restrict__ gh, const _Float16* __restrict__ winf,
            const float* __restrict__ b_in, float* __restrict__ h, int N) {
  int lane = threadIdx.x & 31;
  int wave = threadIdx.x >> 5;
  int tb = blockIdx.x * 128 + wave * 16;
  if (tb >= N) return;   // whole-wave uniform exit; WMMA below runs with full EXEC

  v8f acc[8] = {};
#pragma unroll
  for (int kt = 0; kt < 4; ++kt) {
    v16h a = load_a_frag(gh, HID, tb, kt * 32, lane, N - 1);
#pragma unroll
    for (int nt = 0; nt < 8; ++nt) {
      v16h b = *(const v16h*)(winf + ((size_t)(nt * 4 + kt) * 32 + lane) * 16);
      acc[nt] = wmma_f16(a, b, acc[nt]);
    }
  }
  int rb = tb + ((lane >> 4) << 3);
#pragma unroll
  for (int nt = 0; nt < 8; ++nt) {
    int col = nt * 16 + (lane & 15);
    float bias = b_in[col];
#pragma unroll
    for (int v = 0; v < 8; ++v) {
      int row = rb + v;
      if (row < N) h[(size_t)row * HID + col] = tanhf(acc[nt][v] + bias);
    }
  }
}

// ---------------------------------------------------------------------------
// Fused MLP step: diff = tanh(gelu(aggh@W1+b1)@W2+b2); h = h*(1-decay*dt) + diff*dt.
// Hidden (256 cols) processed in 8 chunks of 32 so GEMM1 output feeds GEMM2's
// K dimension through per-wave LDS staging (C-frag -> A-frag relayout).
// ---------------------------------------------------------------------------
__global__ void __launch_bounds__(256)
mlp_step_k(const _Float16* __restrict__ aggh, const _Float16* __restrict__ w1f,
           const _Float16* __restrict__ w2f, const float* __restrict__ b1,
           const float* __restrict__ b2, const float* __restrict__ clearance,
           float* __restrict__ h, int N) {
  __shared__ __align__(16) _Float16 hstage[8][16 * 32];
  int lane = threadIdx.x & 31;
  int wave = threadIdx.x >> 5;
  int tb = blockIdx.x * 128 + wave * 16;
  if (tb >= N) return;

  const float decay = fmaxf(clearance[0], 0.0f);
  const float keep  = 1.0f - decay * DT;

  v8f outacc[8] = {};
  _Float16* st = &hstage[wave][0];

#pragma unroll 1
  for (int ch = 0; ch < 8; ++ch) {           // hidden cols [32*ch, 32*ch+32)
    // ---- GEMM1: 16x32 chunk of gelu(agg@W1+b1) ----
    v8f hc0 = {}, hc1 = {};
#pragma unroll
    for (int kt = 0; kt < 4; ++kt) {
      v16h a = load_a_frag(aggh, HID, tb, kt * 32, lane, N - 1);
      v16h b0 = *(const v16h*)(w1f + ((size_t)((2 * ch + 0) * 4 + kt) * 32 + lane) * 16);
      v16h b1v = *(const v16h*)(w1f + ((size_t)((2 * ch + 1) * 4 + kt) * 32 + lane) * 16);
      hc0 = wmma_f16(a, b0, hc0);
      hc1 = wmma_f16(a, b1v, hc1);
    }
    // ---- epilogue: exact GELU, stage as f16 A-operand in LDS ----
    int rbase = (lane >> 4) << 3;
    int nloc  = lane & 15;
#pragma unroll
    for (int t = 0; t < 2; ++t) {
      float bias = b1[ch * 32 + t * 16 + nloc];
#pragma unroll
      for (int v = 0; v < 8; ++v) {
        float x = (t == 0 ? hc0[v] : hc1[v]) + bias;
        float g = 0.5f * x * (1.0f + erff(x * 0.70710678118f));
        st[(rbase + v) * 32 + t * 16 + nloc] = (_Float16)g;
      }
    }
    asm volatile("s_wait_dscnt 0" ::: "memory");   // wave-local LDS fence

    // ---- relayout to A-fragment and GEMM2 K-chunk ----
    int m  = lane & 15;
    int kb = (lane >> 4) << 3;
    v8h lo = *(const v8h*)(st + m * 32 + kb);
    v8h hi = *(const v8h*)(st + m * 32 + kb + 16);
    v16h a2 = combine16(lo, hi);
#pragma unroll
    for (int nt = 0; nt < 8; ++nt) {
      v16h b = *(const v16h*)(w2f + ((size_t)(nt * 8 + ch) * 32 + lane) * 16);
      outacc[nt] = wmma_f16(a2, b, outacc[nt]);
    }
    asm volatile("s_wait_dscnt 0" ::: "memory");   // loads drain before next chunk's stores
  }

  // ---- final epilogue: tanh + in-place Euler update ----
  int rb = tb + ((lane >> 4) << 3);
#pragma unroll
  for (int nt = 0; nt < 8; ++nt) {
    int col = nt * 16 + (lane & 15);
    float bias = b2[col];
#pragma unroll
    for (int v = 0; v < 8; ++v) {
      int row = rb + v;
      if (row < N) {
        float diff = tanhf(outacc[nt][v] + bias);
        size_t idx = (size_t)row * HID + col;
        h[idx] = h[idx] * keep + diff * DT;
      }
    }
  }
}

// ---------------------------------------------------------------------------
// Host launcher
// ---------------------------------------------------------------------------
extern "C" void kernel_launch(void* const* d_in, const int* in_sizes, int n_in,
                              void* d_out, int out_size, void* d_ws, size_t ws_size,
                              hipStream_t stream) {
  const float* gat   = (const float*)d_in[0];
  const int*   eidx  = (const int*)d_in[1];   // (2, E) int32 (jax canonicalizes int64->int32)
  const float* W_in  = (const float*)d_in[2];
  const float* b_in  = (const float*)d_in[3];
  const float* W1    = (const float*)d_in[4];
  const float* b1    = (const float*)d_in[5];
  const float* W2    = (const float*)d_in[6];
  const float* b2    = (const float*)d_in[7];
  const float* clear = (const float*)d_in[8];
  float* h = (float*)d_out;                   // d_out doubles as the evolving state h

  const int N = in_sizes[0] / HID;
  const int E = in_sizes[1] / 2;
  const int* src = eidx;
  const int* dst = eidx + E;

  // workspace carve-out (256B aligned)
  char* ws = (char*)d_ws;
  size_t off = 0;
  auto carve = [&](size_t bytes) -> void* {
    void* p = ws + off;
    off = (off + bytes + 255) & ~(size_t)255;
    return p;
  };
  float*     agg    = (float*)    carve((size_t)N * HID * sizeof(float));
  _Float16*  aggh   = (_Float16*) carve((size_t)N * HID * sizeof(_Float16));
  _Float16*  gh     = (_Float16*) carve((size_t)N * HID * sizeof(_Float16));
  _Float16*  winf   = (_Float16*) carve((size_t)HID * HID * sizeof(_Float16));
  _Float16*  w1f    = (_Float16*) carve((size_t)HID * HID2 * sizeof(_Float16));
  _Float16*  w2f    = (_Float16*) carve((size_t)HID2 * HID * sizeof(_Float16));
  float*     invdeg = (float*)    carve((size_t)N * sizeof(float));

  const int TPB = 256;
  auto nb = [](long long n, int tpb) { return (int)((n + tpb - 1) / tpb); };

  // 1) weight / input repack to f16
  prep_weights_k<<<nb(HID * HID,  TPB), TPB, 0, stream>>>(W_in, winf, HID,  HID);
  prep_weights_k<<<nb(HID * HID2, TPB), TPB, 0, stream>>>(W1,   w1f,  HID,  HID2);
  prep_weights_k<<<nb(HID2 * HID, TPB), TPB, 0, stream>>>(W2,   w2f,  HID2, HID);
  cvt_f32_to_f16_k<<<nb((long long)N * HID, TPB), TPB, 0, stream>>>(gat, gh, N * HID);

  // 2) degree -> 1/max(deg,1)
  zero_f32_k<<<nb(N, TPB), TPB, 0, stream>>>(invdeg, N);
  deg_k<<<nb(E, TPB), TPB, 0, stream>>>(dst, invdeg, E);
  invdeg_k<<<nb(N, TPB), TPB, 0, stream>>>(invdeg, N);

  // 3) h0 = tanh(gat @ W_in + b_in)   (WMMA)
  const int gemm_blocks = (N + 127) / 128;
  init_gemm_k<<<gemm_blocks, TPB, 0, stream>>>(gh, winf, b_in, h, N);

  // 4) 12 Euler steps
  for (int s = 0; s < NSTEP; ++s) {
    zero_f32_k<<<nb((long long)N * HID, TPB), TPB, 0, stream>>>(agg, N * HID);
    scatter_k<<<nb((long long)E * 32, TPB), TPB, 0, stream>>>(h, src, dst, agg, E);
    scale_cvt_k<<<nb((long long)N * HID, TPB), TPB, 0, stream>>>(agg, invdeg, aggh, N * HID);
    mlp_step_k<<<gemm_blocks, TPB, 0, stream>>>(aggh, w1f, w2f, b1, b2, clear, h, N);
  }
}
